// Path_Embedding_21139829030933
// MI455X (gfx1250) — compile-verified, matrix operand order checked by reading, not run on hardware
//
#include <hip/hip_runtime.h>
#include <hip/hip_bf16.h>

typedef __attribute__((ext_vector_type(16))) _Float16 v16h;
typedef __attribute__((ext_vector_type(8)))  float    v8f;

#define NB 128
#define NI 64
#define NP 16
#define NTm 5
#define VOCAB 100000
#define ND 64
#define NPAIR (NB * NI * NP)   /* 131072 */
#define NTILES (NPAIR / 4)     /* 32768 wave-tiles of 16 windows */

__global__ void init_out_kernel(float* out, int n) {
  int i = blockIdx.x * blockDim.x + threadIdx.x;
  if (i < n) out[i] = -INFINITY;
}

// Native CDNA5 non-returning float atomic max; RMW resolves at L2 (device
// coherence point). Order-independent => deterministic across replays.
__device__ __forceinline__ void atom_max_f32(float* p, float v) {
  asm volatile("global_atomic_max_num_f32 %0, %1, off scope:SCOPE_DEV"
               :: "v"(p), "v"(v)
               : "memory");
}

__global__ __launch_bounds__(256)
void path_emb_conv_pool(const int* __restrict__ path_input,   // [NPAIR][5]
                        const int* __restrict__ path_type,    // [5]
                        const float* __restrict__ tables,     // [3][VOCAB][64]
                        const float* __restrict__ conv_w,     // [64][64][2]
                        const float* __restrict__ conv_b,     // [64]
                        float* __restrict__ out) {            // [8192][64], pre-set -inf
  const int lane = threadIdx.x & 31;
  const bool hi  = lane >= 16;
  const int mrow = lane & 15;          // M row / N col owned by this lane
  const int q    = mrow >> 2;          // which (n,p) pair within tile (0..3)
  const int tw   = mrow & 3;           // which conv window (0..3)
  const int nwaves = (gridDim.x * blockDim.x) >> 5;
  const int wave   = (blockIdx.x * blockDim.x + threadIdx.x) >> 5;

  int ty[NTm];
  #pragma unroll
  for (int j = 0; j < NTm; ++j) ty[j] = path_type[j];

  // ---- resident B fragments (conv weights as [K=128][O=64], f16) ----
  // f16 16x16x32 layout: lane owns col O = nt*16+mrow; element e<8 -> K=base+e,
  // e>=8 -> K=base+16+(e-8); base = kstep*32 + (hi?8:0).
  // K<64 -> (i=K,   k=0); K>=64 -> (i=K-64, k=1); conv_w[o][i][k] = conv_w[o*128+i*2+k]
  v16h Bf[4][4];
  #pragma unroll
  for (int k4 = 0; k4 < 4; ++k4) {
    const int base = k4 * 32 + (hi ? 8 : 0);
    #pragma unroll
    for (int nt = 0; nt < 4; ++nt) {
      const int o = nt * 16 + mrow;
      v16h b;
      #pragma unroll
      for (int j = 0; j < 8; ++j) {
        const int K1 = base + j, K2 = base + 16 + j;
        b[j]     = (_Float16)conv_w[o * 128 + (K1 & 63) * 2 + (K1 >> 6)];
        b[8 + j] = (_Float16)conv_w[o * 128 + (K2 & 63) * 2 + (K2 >> 6)];
      }
      Bf[k4][nt] = b;
    }
  }
  float biasv[4];
  #pragma unroll
  for (int nt = 0; nt < 4; ++nt) biasv[nt] = conv_b[nt * 16 + mrow];

  for (int tile = wave; tile < NTILES; tile += nwaves) {
    const int pairBase = tile << 2;
    const int myPair   = pairBase + q;
    // this lane's window (q, tw) needs embedding rows at times tw and tw+1
    const float* rp[2];
    #pragma unroll
    for (int s = 0; s < 2; ++s) {
      const int tt  = tw + s;
      const int idx = path_input[myPair * NTm + tt];
      rp[s] = tables + ((long)ty[tt] * VOCAB + idx) * ND;
    }

    v8f acc[4] = {};
    #pragma unroll
    for (int k4 = 0; k4 < 4; ++k4) {
      // A fragment: window row = [emb(t, 0..63), emb(t+1, 0..63)];
      // kstep k4 covers K = k4*32..k4*32+31 -> time = tw + (k4>=2),
      // cols = (k4&1)*32 + (hi?8:0) + {0..7, 16..23}
      const float* r  = rp[k4 >> 1];
      const int    c0 = ((k4 & 1) << 5) + (hi ? 8 : 0);
      v16h a;
      #pragma unroll
      for (int j = 0; j < 8; ++j) {
        a[j]     = (_Float16)r[c0 + j];
        a[8 + j] = (_Float16)r[c0 + 16 + j];
      }
      #pragma unroll
      for (int nt = 0; nt < 4; ++nt)
        acc[nt] = __builtin_amdgcn_wmma_f32_16x16x32_f16(
            false, a, false, Bf[k4][nt], (short)0, acc[nt], false, false);
    }

    // ---- epilogue: time-max (+bias), then fused 16-way strided max via atomics ----
    // C layout: VGPR r holds row M=r (lanes 0-15) / M=r+8 (lanes 16-31), col = mrow.
    // Rows group as q*4+t, so max over 4 consecutive accumulator VGPRs = time pool.
    const int qlo   = hi ? 2 : 0;
    const int pair0 = pairBase + qlo;
    const int pair1 = pair0 + 1;
    // pooled[n0=pair>>4, p0=pair&15] contributes to out row (n0%512)*16 + p0
    const int nn0 = (((pair0 >> 4) & 511) << 4) + (pair0 & 15);
    const int nn1 = (((pair1 >> 4) & 511) << 4) + (pair1 & 15);
    #pragma unroll
    for (int nt = 0; nt < 4; ++nt) {
      const v8f c = acc[nt];
      const float m0 = fmaxf(fmaxf(c[0], c[1]), fmaxf(c[2], c[3])) + biasv[nt];
      const float m1 = fmaxf(fmaxf(c[4], c[5]), fmaxf(c[6], c[7])) + biasv[nt];
      const int o = nt * 16 + mrow;
      atom_max_f32(&out[nn0 * 64 + o], m0);
      atom_max_f32(&out[nn1 * 64 + o], m1);
    }
  }
}

extern "C" void kernel_launch(void* const* d_in, const int* in_sizes, int n_in,
                              void* d_out, int out_size, void* d_ws, size_t ws_size,
                              hipStream_t stream) {
  const int*   path_input = (const int*)d_in[0];
  const int*   path_type  = (const int*)d_in[1];
  const float* tables     = (const float*)d_in[2];
  const float* conv_w     = (const float*)d_in[3];
  const float* conv_b     = (const float*)d_in[4];
  float* out = (float*)d_out;

  init_out_kernel<<<(out_size + 255) / 256, 256, 0, stream>>>(out, out_size);
  path_emb_conv_pool<<<1024, 256, 0, stream>>>(path_input, path_type, tables,
                                               conv_w, conv_b, out);
}
